// GPTLanguageModel_15564961480824
// MI455X (gfx1250) — matrix operand order repaired and checked by reading, not compiled, not run on hardware
//
#include <hip/hip_runtime.h>

typedef __attribute__((ext_vector_type(16))) __bf16   v16bf;
typedef __attribute__((ext_vector_type(8)))  float    v8f;
typedef __attribute__((ext_vector_type(4)))  unsigned u32x4;

#define NL 6
#define NE 384
#define NV 32000
#define NH 6
#define ND 64
#define NF 1536
#define NB 32
#define NT 256
#define NM (NB * NT)   /* 8192 token rows */

// Issue two async 16B global->LDS copies (32 contiguous bytes). INST_OFFSET is
// applied to BOTH the global and the LDS address (ISA 15.18.3), so a single
// address pair covers both halves. Tracked by ASYNCcnt.
#define ASYNC_LDS_COPY32(ldsoff, gaddr)                                        \
    asm volatile("global_load_async_to_lds_b128 %0, %1, off\n\t"               \
                 "global_load_async_to_lds_b128 %0, %1, off offset:16"         \
                 :: "v"(ldsoff), "v"(gaddr) : "memory")

__device__ __forceinline__ unsigned lds_offset_of(const void* p)
{
    // Generic (flat) LDS address: low 32 bits are the wave-relative LDS offset.
    return (unsigned)(unsigned long long)p;
}

// ---------------------------------------------------------------------------
// Generic batched bf16 WMMA GEMM: C = op(A @ B [+ bias] [+ res]) per batch z.
// A: [M,K] bf16 row-major (lda). B: weights [K,N] bf16 (BT=0) or [N,K] (BT=1).
// Tile: 128x128 per block, 8 waves, each wave = 32(M) x 64(N) = 2x4 WMMA tiles.
// NOTE: async staging assumes M % 128 == 0 and (BT ? N % 128 == 0 : true),
// which holds for every launch below (M in {8192,256}, BT only with N=256).
// ---------------------------------------------------------------------------
template<bool BT, bool CAUSAL, bool BIAS, bool RES, bool RELU, bool OUTBF16, bool NTSTORE>
__global__ __launch_bounds__(256)
void gemm_wmma(const __bf16* __restrict__ A, const __bf16* __restrict__ Bm,
               void* __restrict__ Cv, const float* __restrict__ bias,
               const float* __restrict__ res,
               int M, int N, int K, int lda, int ldb, int ldc,
               int batchDiv,
               long long aS1, long long aS2, long long bS1, long long bS2,
               long long cS1, long long cS2)
{
    constexpr int LDT = 40;                 // padded LDS row stride (bf16 elems)
    __shared__ __bf16 As[128 * LDT];
    __shared__ __bf16 Bs[128 * LDT];

    const int z  = blockIdx.z;
    const int z1 = z / batchDiv;
    const int z2 = z % batchDiv;
    A  += z1 * aS1 + z2 * aS2;
    Bm += z1 * bS1 + z2 * bS2;
    const long long cOff = z1 * cS1 + z2 * cS2;

    const int n0 = blockIdx.x * 128;
    const int m0 = blockIdx.y * 128;
    const int tid  = threadIdx.x;
    const int lane = tid & 31;
    const int wave = tid >> 5;
    const int wm = wave >> 1, wn = wave & 1;     // 4 x 2 wave grid
    const int lh = lane >> 4, l16 = lane & 15;

    v8f c[2][4];
#pragma unroll
    for (int i = 0; i < 2; ++i)
#pragma unroll
        for (int j = 0; j < 4; ++j)
#pragma unroll
            for (int r = 0; r < 8; ++r) c[i][j][r] = 0.0f;

    // Per-thread staging coordinates (fixed across k-steps).
    const int arow = tid >> 1, ahalf = tid & 1;
    const unsigned aLds = lds_offset_of(&As[arow * LDT + ahalf * 16]);
    const unsigned bLdsT = lds_offset_of(&Bs[arow * LDT + ahalf * 16]);

    const int nk = K >> 5;
    for (int kt = 0; kt < nk; ++kt) {
        __syncthreads();
        // ---- stage A tile (128 x 32) via async global->LDS copies ---------
        {
            const unsigned long long ga = (unsigned long long)
                (A + (long long)(m0 + arow) * lda + kt * 32 + ahalf * 16);
            ASYNC_LDS_COPY32(aLds, ga);
            if (kt + 1 < nk)
                __builtin_prefetch((const void*)(A + (long long)(m0 + arow) * lda
                                                 + (kt + 1) * 32 + ahalf * 16), 0, 1);
        }
        // ---- stage B tile into N-major LDS [n][k] -------------------------
        if (BT) {
            // B source is [N,K] row-major (K-matrix for scores): direct async copy
            const unsigned long long gb = (unsigned long long)
                (Bm + (long long)(n0 + arow) * ldb + kt * 32 + ahalf * 16);
            ASYNC_LDS_COPY32(bLdsT, gb);
        } else {
            // B source is [K,N] row-major (weights): transpose on LDS store
            const int k = tid >> 3, nn = (tid & 7) * 16;
            const int gn = n0 + nn;
            union { u32x4 u[2]; __bf16 h[16]; } t;
            t.u[0] = (u32x4){0,0,0,0}; t.u[1] = (u32x4){0,0,0,0};
            if (gn < N) {
                const u32x4* src = (const u32x4*)(Bm + (long long)(kt * 32 + k) * ldb + gn);
                t.u[0] = src[0]; t.u[1] = src[1];
                if (kt + 1 < nk)
                    __builtin_prefetch((const void*)(Bm + (long long)((kt + 1) * 32 + k) * ldb + gn), 0, 1);
            }
#pragma unroll
            for (int i = 0; i < 16; ++i) Bs[(nn + i) * LDT + k] = t.h[i];
        }
        asm volatile("s_wait_asynccnt 0x0" ::: "memory");
        __syncthreads();

        // ---- fragments (contiguous k: lanes 0-15 K0..15,
        //      lanes 16-31 K16..31; same for A and B fragments) -
        union Frag { u32x4 u[2]; v16bf v; };
        v16bf af[2];
#pragma unroll
        for (int mt = 0; mt < 2; ++mt) {
            const int row = wm * 32 + mt * 16 + l16;
            const u32x4* p = (const u32x4*)&As[row * LDT + lh * 16];
            Frag f; f.u[0] = p[0]; f.u[1] = p[1];
            af[mt] = f.v;
        }
        v16bf bf[4];
#pragma unroll
        for (int nt = 0; nt < 4; ++nt) {
            const int col = wn * 64 + nt * 16 + l16;
            const u32x4* p = (const u32x4*)&Bs[col * LDT + lh * 16];
            Frag f; f.u[0] = p[0]; f.u[1] = p[1];
            bf[nt] = f.v;
        }
#pragma unroll
        for (int mt = 0; mt < 2; ++mt)
#pragma unroll
            for (int nt = 0; nt < 4; ++nt)
                c[mt][nt] = __builtin_amdgcn_wmma_f32_16x16x32_bf16(
                    false, af[mt], false, bf[nt], (short)0, c[mt][nt], false, false);
    }

    // ---- epilogue ---------------------------------------------------------
#pragma unroll
    for (int mt = 0; mt < 2; ++mt) {
#pragma unroll
        for (int nt = 0; nt < 4; ++nt) {
            const int n = n0 + wn * 64 + nt * 16 + l16;
            if (n >= N) continue;
            const float bv = BIAS ? bias[n] : 0.0f;
#pragma unroll
            for (int r = 0; r < 8; ++r) {
                const int m = m0 + wm * 32 + mt * 16 + lh * 8 + r;
                if (m >= M) continue;
                float v = c[mt][nt][r];
                if (CAUSAL) { v *= 0.125f; if (n > m) v = -1e30f; }
                if (BIAS) v += bv;
                if (RES)  v += res[cOff + (long long)m * ldc + n];
                if (RELU) v = fmaxf(v, 0.0f);
                const long long idx = cOff + (long long)m * ldc + n;
                if (OUTBF16)      ((__bf16*)Cv)[idx] = (__bf16)v;
                else if (NTSTORE) __builtin_nontemporal_store(v, &((float*)Cv)[idx]);
                else              ((float*)Cv)[idx]  = v;
            }
        }
    }
}

// ---------------------------------------------------------------------------
// LayerNorm: one wave (32 lanes) per 384-elem row, f32 in -> bf16 out.
// ---------------------------------------------------------------------------
__global__ __launch_bounds__(256)
void ln_bf16_kernel(const float* __restrict__ x, const float* __restrict__ g,
                    const float* __restrict__ b, __bf16* __restrict__ out, int rows)
{
    const int wave = threadIdx.x >> 5, lane = threadIdx.x & 31;
    const int row = blockIdx.x * 8 + wave;
    if (row >= rows) return;
    const float* xr = x + (long long)row * NE;
    float v[12], s = 0.0f, sq = 0.0f;
#pragma unroll
    for (int i = 0; i < 12; ++i) { v[i] = xr[lane + i * 32]; s += v[i]; sq += v[i] * v[i]; }
#pragma unroll
    for (int m = 16; m >= 1; m >>= 1) { s += __shfl_xor(s, m, 32); sq += __shfl_xor(sq, m, 32); }
    const float mean = s * (1.0f / NE);
    const float var  = sq * (1.0f / NE) - mean * mean;
    const float rstd = rsqrtf(var + 1e-5f);
    __bf16* o = out + (long long)row * NE;
#pragma unroll
    for (int i = 0; i < 12; ++i) {
        const int cidx = lane + i * 32;
        o[cidx] = (__bf16)((v[i] - mean) * rstd * g[cidx] + b[cidx]);
    }
}

// ---------------------------------------------------------------------------
// Softmax over T=256 keys: one wave per row, f32 scores -> bf16 probs.
// ---------------------------------------------------------------------------
__global__ __launch_bounds__(256)
void softmax_bf16_kernel(const float* __restrict__ s, __bf16* __restrict__ p, int rows)
{
    const int wave = threadIdx.x >> 5, lane = threadIdx.x & 31;
    const int row = blockIdx.x * 8 + wave;
    if (row >= rows) return;
    const float* sr = s + (long long)row * NT;
    float v[8], mx = -3.0e38f;
#pragma unroll
    for (int i = 0; i < 8; ++i) { v[i] = sr[lane + i * 32]; mx = fmaxf(mx, v[i]); }
#pragma unroll
    for (int m = 16; m >= 1; m >>= 1) mx = fmaxf(mx, __shfl_xor(mx, m, 32));
    float sum = 0.0f;
#pragma unroll
    for (int i = 0; i < 8; ++i) { v[i] = __expf(v[i] - mx); sum += v[i]; }
#pragma unroll
    for (int m = 16; m >= 1; m >>= 1) sum += __shfl_xor(sum, m, 32);
    const float inv = 1.0f / sum;
    __bf16* pr = p + (long long)row * NT;
#pragma unroll
    for (int i = 0; i < 8; ++i) pr[lane + i * 32] = (__bf16)(v[i] * inv);
}

// ---------------------------------------------------------------------------
__global__ void embed_kernel(const int* __restrict__ tok, const float* __restrict__ te,
                             const float* __restrict__ pe, float* __restrict__ x)
{
    const long long i = (long long)blockIdx.x * 256 + threadIdx.x;
    if (i >= (long long)NM * NE) return;
    const int cidx = (int)(i % NE);
    const long long bt = i / NE;
    const int t = (int)(bt % NT);
    const int tk = tok[bt];
    x[i] = te[(long long)tk * NE + cidx] + pe[t * NE + cidx];
}

__global__ void cvt_bf16_kernel(const float* __restrict__ src, __bf16* __restrict__ dst,
                                long long n)
{
    const long long i = (long long)blockIdx.x * 256 + threadIdx.x;
    if (i < n) dst[i] = (__bf16)src[i];
}

// ---------------------------------------------------------------------------
// Loss: one-pass online logsumexp per row (saves a full 1GB re-read of the
// logits), non-temporal loads, then deterministic block sum.
// ---------------------------------------------------------------------------
__global__ __launch_bounds__(256)
void loss_rows_kernel(const float* __restrict__ logits, const int* __restrict__ tgt,
                      float* __restrict__ rowloss)
{
    __shared__ float rm[8], rs[8];
    const int row = blockIdx.x;
    const float* lr = logits + (long long)row * NV;
    const int tid = threadIdx.x, lane = tid & 31, wave = tid >> 5;

    float m = -3.4e38f, s = 0.0f;
    for (int i = tid; i < NV; i += 256) {
        const float xv = __builtin_nontemporal_load(&lr[i]);
        const float mn = fmaxf(m, xv);
        s = s * __expf(m - mn) + __expf(xv - mn);
        m = mn;
    }
#pragma unroll
    for (int d = 16; d >= 1; d >>= 1) {
        const float om = __shfl_xor(m, d, 32);
        const float os = __shfl_xor(s, d, 32);
        const float mn = fmaxf(m, om);
        s = s * __expf(m - mn) + os * __expf(om - mn);
        m = mn;
    }
    if (lane == 0) { rm[wave] = m; rs[wave] = s; }
    __syncthreads();
    if (tid == 0) {
        float M = rm[0], S = rs[0];
        for (int i = 1; i < 8; ++i) {
            const float mn = fmaxf(M, rm[i]);
            S = S * __expf(M - mn) + rs[i] * __expf(rm[i] - mn);
            M = mn;
        }
        rowloss[row] = (__logf(S) + M) - lr[tgt[row]];
    }
}

__global__ __launch_bounds__(256)
void loss_final_kernel(const float* __restrict__ rowloss, float* __restrict__ out)
{
    __shared__ float red[256];
    float s = 0.0f;
    for (int i = threadIdx.x; i < NM; i += 256) s += rowloss[i];
    red[threadIdx.x] = s;
    __syncthreads();
    for (int st = 128; st > 0; st >>= 1) {
        if (threadIdx.x < st) red[threadIdx.x] += red[threadIdx.x + st];
        __syncthreads();
    }
    if (threadIdx.x == 0) out[0] = red[0] / (float)NM;
}

// ---------------------------------------------------------------------------
extern "C" void kernel_launch(void* const* d_in, const int* in_sizes, int n_in,
                              void* d_out, int out_size, void* d_ws, size_t ws_size,
                              hipStream_t stream)
{
    (void)in_sizes; (void)n_in; (void)out_size; (void)ws_size;
    const int*   tokens  = (const int*)d_in[0];
    const int*   targets = (const int*)d_in[1];
    const float* tok_emb = (const float*)d_in[2];
    const float* pos_emb = (const float*)d_in[3];
    const float* wq = (const float*)d_in[4];
    const float* wk = (const float*)d_in[5];
    const float* wv = (const float*)d_in[6];
    const float* wp = (const float*)d_in[7];
    const float* bp = (const float*)d_in[8];
    const float* w1 = (const float*)d_in[9];
    const float* b1 = (const float*)d_in[10];
    const float* w2 = (const float*)d_in[11];
    const float* b2 = (const float*)d_in[12];
    const float* ln1g = (const float*)d_in[13];
    const float* ln1b = (const float*)d_in[14];
    const float* ln2g = (const float*)d_in[15];
    const float* ln2b = (const float*)d_in[16];
    const float* lnfg = (const float*)d_in[17];
    const float* lnfb = (const float*)d_in[18];
    const float* wlm  = (const float*)d_in[19];
    const float* blm  = (const float*)d_in[20];

    char* ws = (char*)d_ws; size_t off = 0;
    auto alloc = [&](size_t bytes) -> void* {
        void* p = ws + off; off = (off + bytes + 255) & ~(size_t)255; return p;
    };
    float*  x   = (float*)alloc((size_t)NM * NE * 4);
    __bf16* hb  = (__bf16*)alloc((size_t)NM * NE * 2);
    __bf16* qb  = (__bf16*)alloc((size_t)NM * NE * 2);
    __bf16* kb  = (__bf16*)alloc((size_t)NM * NE * 2);
    __bf16* vb  = (__bf16*)alloc((size_t)NM * NE * 2);
    __bf16* ob  = (__bf16*)alloc((size_t)NM * NE * 2);
    float*  sc  = (float*)alloc((size_t)NB * NH * NT * NT * 4);
    __bf16* pb  = (__bf16*)alloc((size_t)NB * NH * NT * NT * 2);
    __bf16* h1  = (__bf16*)alloc((size_t)NM * NF * 2);
    __bf16* wqb = (__bf16*)alloc((size_t)NL * NE * NE * 2);
    __bf16* wkb = (__bf16*)alloc((size_t)NL * NE * NE * 2);
    __bf16* wvb = (__bf16*)alloc((size_t)NL * NE * NE * 2);
    __bf16* wpb = (__bf16*)alloc((size_t)NL * NE * NE * 2);
    __bf16* w1b = (__bf16*)alloc((size_t)NL * NE * NF * 2);
    __bf16* w2b = (__bf16*)alloc((size_t)NL * NF * NE * 2);
    __bf16* wlmb = (__bf16*)alloc((size_t)NE * NV * 2);
    float*  rowloss = (float*)alloc((size_t)NM * 4);

    // --- weight conversion f32 -> bf16 (once per call) ---
    auto cvt = [&](const float* s, __bf16* d, long long n) {
        cvt_bf16_kernel<<<(unsigned)((n + 255) / 256), 256, 0, stream>>>(s, d, n);
    };
    cvt(wq, wqb, (long long)NL * NE * NE);
    cvt(wk, wkb, (long long)NL * NE * NE);
    cvt(wv, wvb, (long long)NL * NE * NE);
    cvt(wp, wpb, (long long)NL * NE * NE);
    cvt(w1, w1b, (long long)NL * NE * NF);
    cvt(w2, w2b, (long long)NL * NF * NE);
    cvt(wlm, wlmb, (long long)NE * NV);

    // --- embedding ---
    embed_kernel<<<(unsigned)(((long long)NM * NE + 255) / 256), 256, 0, stream>>>(
        tokens, tok_emb, pos_emb, x);

    const dim3 gProj(3, NM / 128, 1);
    const dim3 gF1(NF / 128, NM / 128, 1);
    const dim3 gScore(2, 2, NB * NH);
    const dim3 gAV(1, 2, NB * NH);
    const dim3 gLM(NV / 128, NM / 128, 1);

    for (int l = 0; l < NL; ++l) {
        const __bf16* lwq = wqb + (long long)l * NE * NE;
        const __bf16* lwk = wkb + (long long)l * NE * NE;
        const __bf16* lwv = wvb + (long long)l * NE * NE;
        const __bf16* lwp = wpb + (long long)l * NE * NE;
        const __bf16* lw1 = w1b + (long long)l * NE * NF;
        const __bf16* lw2 = w2b + (long long)l * NF * NE;

        // LN1 -> hb (bf16)
        ln_bf16_kernel<<<NM / 8, 256, 0, stream>>>(x, ln1g + l * NE, ln1b + l * NE, hb, NM);

        // q/k/v = hb @ W  (bf16 out)
        gemm_wmma<false,false,false,false,false,true,false><<<gProj, 256, 0, stream>>>(
            hb, lwq, qb, nullptr, nullptr, NM, NE, NE, NE, NE, NE, 1, 0,0,0,0,0,0);
        gemm_wmma<false,false,false,false,false,true,false><<<gProj, 256, 0, stream>>>(
            hb, lwk, kb, nullptr, nullptr, NM, NE, NE, NE, NE, NE, 1, 0,0,0,0,0,0);
        gemm_wmma<false,false,false,false,false,true,false><<<gProj, 256, 0, stream>>>(
            hb, lwv, vb, nullptr, nullptr, NM, NE, NE, NE, NE, NE, 1, 0,0,0,0,0,0);

        // scores = causal(0.125 * q @ k^T) per (b,h): batched, BT mode
        gemm_wmma<true,true,false,false,false,false,false><<<gScore, 256, 0, stream>>>(
            qb, kb, sc, nullptr, nullptr, NT, NT, ND, NE, NE, NT, NH,
            (long long)NT * NE, ND, (long long)NT * NE, ND,
            (long long)NH * NT * NT, (long long)NT * NT);

        // softmax -> probs bf16
        softmax_bf16_kernel<<<(NB * NH * NT) / 8, 256, 0, stream>>>(sc, pb, NB * NH * NT);

        // o = P @ V per (b,h): bf16 out into [B,T,E] layout
        gemm_wmma<false,false,false,false,false,true,false><<<gAV, 256, 0, stream>>>(
            pb, vb, ob, nullptr, nullptr, NT, ND, NT, NT, NE, NE, NH,
            (long long)NH * NT * NT, (long long)NT * NT, (long long)NT * NE, ND,
            (long long)NT * NE, ND);

        // x = x + o @ wproj + bproj
        gemm_wmma<false,false,true,true,false,false,false><<<gProj, 256, 0, stream>>>(
            ob, lwp, x, bp + l * NE, x, NM, NE, NE, NE, NE, NE, 1, 0,0,0,0,0,0);

        // LN2 -> hb
        ln_bf16_kernel<<<NM / 8, 256, 0, stream>>>(x, ln2g + l * NE, ln2b + l * NE, hb, NM);

        // h1 = relu(hb @ w1 + b1)  (bf16 out)
        gemm_wmma<false,false,true,false,true,true,false><<<gF1, 256, 0, stream>>>(
            hb, lw1, h1, b1 + l * NF, nullptr, NM, NF, NE, NE, NF, NF, 1, 0,0,0,0,0,0);

        // x = x + h1 @ w2 + b2
        gemm_wmma<false,false,true,true,false,false,false><<<gProj, 256, 0, stream>>>(
            h1, lw2, x, b2 + l * NE, x, NM, NE, NF, NF, NE, NE, 1, 0,0,0,0,0,0);
    }

    // final LN -> hb
    ln_bf16_kernel<<<NM / 8, 256, 0, stream>>>(x, lnfg, lnfb, hb, NM);

    // logits = hb @ w_lm + b_lm  (f32, non-temporal, directly into d_out)
    float* logits = (float*)d_out;
    gemm_wmma<false,false,true,false,false,false,true><<<gLM, 256, 0, stream>>>(
        hb, wlmb, logits, blm, nullptr, NM, NV, NE, NE, NV, NV, 1, 0,0,0,0,0,0);

    // loss (deterministic two-stage reduction) -> d_out[NM*NV]
    loss_rows_kernel<<<NM, 256, 0, stream>>>(logits, targets, rowloss);
    loss_final_kernel<<<1, 256, 0, stream>>>(rowloss, logits + (long long)NM * NV);
}
